// MixRestormer_38955353374993
// MI455X (gfx1250) — matrix operand order repaired and checked by reading, not compiled, not run on hardware
//
#include <hip/hip_runtime.h>
#include <hip/hip_bf16.h>
#include <math.h>

typedef __attribute__((ext_vector_type(16))) _Float16 v16h;
typedef __attribute__((ext_vector_type(8)))  _Float16 v8h;
typedef __attribute__((ext_vector_type(8)))  float    v8f;

#define H_   192
#define W_   192
#define HW_  36864
#define B_   4

__device__ __forceinline__ float gelu_f(float x) {
  return 0.5f * x * (1.0f + erff(x * 0.70710678118654752f));
}
__device__ __forceinline__ float sigmoid_f(float x) {
  return 1.0f / (1.0f + __expf(-x));
}

// ---------------------------------------------------------------------------
// Generic 1x1-conv GEMM: out[b][co][pix] = sum_k W[co][k]*in[b][k][pix] + bias
// 256 threads = 8 waves. Wave grid: MW waves along M x PW waves along pixels
// (MW*PW == 8). Each wave owns MR M-tiles (16 rows each) -> MR v8f
// accumulators, so a block covers (MW*MR*16) Cout x (PW*16) pixels and the
// LDS weight stage is amortized over MR x more WMMAs.
// epi : 0 none, 1 gelu, 2 gelu+residual, 3 residual, 4 per-pixel scale
// ---------------------------------------------------------------------------
template <int MR>
__global__ __launch_bounds__(256)
void gemm1x1_wmma(const float* __restrict__ in, long long ibs,
                  const float* __restrict__ w, const float* __restrict__ bias,
                  float* __restrict__ out, long long obs,
                  int Cin, int MW, int PW,
                  int epi, const float* __restrict__ extra, long long ebs)
{
  __shared__ __align__(16) _Float16 sA[17408];   // [MW*MR*16 rows][Cin+8]
  __shared__ __align__(16) _Float16 sB[17408];   // [PW*16 cols][Cin+8]
  const int tid   = threadIdx.x;
  const int b     = blockIdx.y;
  const int rows  = MW * MR * 16;
  const int cbase = blockIdx.z * rows;
  const int cols  = PW * 16;
  const int pix0  = blockIdx.x * cols;
  const int SLD   = Cin + 8;                      // pad: keeps 16B alignment
  const float* inB = in + (long long)b * ibs;

  // ---- stage weights (row-major) and activations (col-major, f32->f16) ----
  const int nA = rows * Cin;
  for (int i = tid; i < nA; i += 256) {
    int k = i % Cin, r = i / Cin;
    sA[r * SLD + k] = (_Float16)w[(long long)(cbase + r) * Cin + k];
  }
  const int nB = cols * Cin;
  for (int i = tid; i < nB; i += 256) {
    int col = i % cols, k = i / cols;             // consecutive tid -> coalesced
    sB[col * SLD + k] = (_Float16)inB[(long long)k * HW_ + pix0 + col];
  }
  if (tid == 0)  // hint next pixel-tile into cache (global_prefetch_b8)
    __builtin_prefetch(inB + pix0 + cols, 0, 1);
  __syncthreads();

  const int lane = tid & 31;            // wave32
  const int wv   = tid >> 5;
  const int mw   = wv % MW;
  const int p    = wv / MW;
  const int lm   = lane & 15;
  const int hi   = lane >> 4;

  v8f acc[MR];
  #pragma unroll
  for (int j = 0; j < MR; ++j) acc[j] = (v8f){};

  const _Float16* aBase = &sA[(mw * MR * 16 + lm) * SLD];
  const _Float16* bCol  = &sB[(p * 16 + lm) * SLD];
  for (int kk = 0; kk < Cin; kk += 32) {
    // B frag (32x16 f16 layout): K = kk+16*hi+[0..15] contiguous
    v8h blo = *(const v8h*)(bCol + kk + hi * 16);
    v8h bhi = *(const v8h*)(bCol + kk + hi * 16 + 8);
    v16h bb;
    #pragma unroll
    for (int i = 0; i < 8; ++i) { bb[i] = blo[i]; bb[i + 8] = bhi[i]; }
    #pragma unroll
    for (int j = 0; j < MR; ++j) {
      // A frag (16x32 f16 layout): K = kk+16*hi+[0..15] contiguous
      const _Float16* aRow = aBase + (long long)j * 16 * SLD;
      v8h alo = *(const v8h*)(aRow + kk + hi * 16);
      v8h ahi = *(const v8h*)(aRow + kk + hi * 16 + 8);
      v16h a;
      #pragma unroll
      for (int i = 0; i < 8; ++i) { a[i] = alo[i]; a[i + 8] = ahi[i]; }
      acc[j] = __builtin_amdgcn_wmma_f32_16x16x32_f16(false, a, false, bb,
                                                      (short)0, acc[j], false, false);
    }
  }

  // ---- epilogue: D VGPR r -> M = r + 8*hi, N = lm ----
  const int pix = pix0 + p * 16 + lm;
  #pragma unroll
  for (int j = 0; j < MR; ++j) {
    #pragma unroll
    for (int r = 0; r < 8; ++r) {
      int ch = cbase + (mw * MR + j) * 16 + hi * 8 + r;
      float v = acc[j][r] + bias[ch];
      if      (epi == 1) v = gelu_f(v);
      else if (epi == 2) v = gelu_f(v) + extra[(long long)b * ebs + (long long)ch * HW_ + pix];
      else if (epi == 3) v = v + extra[(long long)b * ebs + (long long)ch * HW_ + pix];
      else if (epi == 4) v = v * extra[(long long)b * ebs + pix];
      out[(long long)b * obs + (long long)ch * HW_ + pix] = v;
    }
  }
}

// --------------------------- LayerNorm2d (+opt GELU) -----------------------
__global__ __launch_bounds__(256)
void ln2d_k(const float* __restrict__ in, float* __restrict__ out,
            const float* __restrict__ g, const float* __restrict__ bt,
            int C, int do_gelu)
{
  long long idx = (long long)blockIdx.x * 256 + threadIdx.x;
  if (idx >= (long long)B_ * HW_) return;
  long long bb = idx / HW_, pix = idx % HW_;
  const float* ip = in + bb * (long long)C * HW_ + pix;
  float s = 0.f, sq = 0.f;
  for (int c = 0; c < C; ++c) { float v = ip[(long long)c * HW_]; s += v; sq += v * v; }
  float mu  = s / (float)C;
  float var = fmaxf(sq / (float)C - mu * mu, 0.f);
  float inv = rsqrtf(var + 1e-6f);
  float* op = out + bb * (long long)C * HW_ + pix;
  for (int c = 0; c < C; ++c) {
    float v = (ip[(long long)c * HW_] - mu) * inv * g[c] + bt[c];
    if (do_gelu) v = gelu_f(v);
    op[(long long)c * HW_] = v;
  }
}

// --------------------------- depthwise 3x3 SAME (+opt GELU) ----------------
__global__ __launch_bounds__(256)
void dwconv3x3_k(const float* __restrict__ in, long long ibs,
                 const float* __restrict__ w, const float* __restrict__ bias,
                 float* __restrict__ out, long long obs, int C, int do_gelu)
{
  long long idx = (long long)blockIdx.x * 256 + threadIdx.x;
  long long tot = (long long)B_ * C * HW_;
  if (idx >= tot) return;
  int b = (int)(idx / ((long long)C * HW_));
  long long rem = idx % ((long long)C * HW_);
  int c = (int)(rem / HW_);
  int p = (int)(rem % HW_);
  int yy = p / W_, xx = p % W_;
  const float* ip = in + (long long)b * ibs + (long long)c * HW_;
  float acc = bias[c];
  #pragma unroll
  for (int ky = -1; ky <= 1; ++ky)
    #pragma unroll
    for (int kx = -1; kx <= 1; ++kx) {
      int y2 = yy + ky, x2 = xx + kx;
      if (y2 >= 0 && y2 < H_ && x2 >= 0 && x2 < W_)
        acc += w[c * 9 + (ky + 1) * 3 + (kx + 1)] * ip[y2 * W_ + x2];
    }
  if (do_gelu) acc = gelu_f(acc);
  out[(long long)b * obs + (long long)c * HW_ + p] = acc;
}

// --------------------------- global mean pool per (b, c<64) ----------------
__global__ __launch_bounds__(256)
void pool_k(const float* __restrict__ in, float* __restrict__ pooled)
{
  __shared__ float red[256];
  int rid = blockIdx.x;                 // b*64 + c
  int b = rid >> 6, c = rid & 63;
  const float* ip = in + (long long)b * 64 * HW_ + (long long)c * HW_;
  float s = 0.f;
  for (int i = threadIdx.x; i < HW_; i += 256) s += ip[i];
  red[threadIdx.x] = s; __syncthreads();
  for (int st = 128; st > 0; st >>= 1) {
    if (threadIdx.x < st) red[threadIdx.x] += red[threadIdx.x + st];
    __syncthreads();
  }
  if (threadIdx.x == 0) pooled[rid] = red[0] / (float)HW_;
}

// --------------------------- squeeze-excite MLP ----------------------------
__global__ __launch_bounds__(256)
void se_k(const float* __restrict__ pooled,
          const float* __restrict__ w1, const float* __restrict__ b1,
          const float* __restrict__ w2, const float* __restrict__ b2,
          float* __restrict__ chia)
{
  __shared__ float hid[64];             // 4 batches * 16 hidden
  int t = threadIdx.x;
  if (t < 64) {
    int b = t >> 4, j = t & 15;
    float s = b1[j];
    for (int c = 0; c < 64; ++c) s += w1[j * 64 + c] * pooled[b * 64 + c];
    hid[t] = gelu_f(s);
  }
  __syncthreads();
  int b = t >> 6, o = t & 63;
  float s = b2[o];
  for (int j = 0; j < 16; ++j) s += w2[o * 16 + j] * hid[b * 16 + j];
  chia[t] = sigmoid_f(s);
}

// --------------------------- 1/max(||row||, eps) for q and k ---------------
__global__ __launch_bounds__(256)
void norm_k(const float* __restrict__ qkvd, float* __restrict__ inv)
{
  __shared__ float red[256];
  int rid = blockIdx.x;                 // [isK(2)][b(4)][c(64)]
  int isK = rid >> 8;
  int b = (rid >> 6) & 3;
  int c = rid & 63;
  const float* ip = qkvd + (long long)b * 192 * HW_ + (long long)(isK * 64 + c) * HW_;
  float s = 0.f;
  for (int i = threadIdx.x; i < HW_; i += 256) { float v = ip[i]; s += v * v; }
  red[threadIdx.x] = s; __syncthreads();
  for (int st = 128; st > 0; st >>= 1) {
    if (threadIdx.x < st) red[threadIdx.x] += red[threadIdx.x + st];
    __syncthreads();
  }
  if (threadIdx.x == 0) inv[rid] = 1.0f / fmaxf(sqrtf(red[0]), 1e-12f);
}

// --------------------------- 8x8 channel attention + softmax ---------------
__global__ __launch_bounds__(256)
void attn_k(const float* __restrict__ qkvd, const float* __restrict__ inv,
            const float* __restrict__ temp, float* __restrict__ attn)
{
  __shared__ float qs[8 * 256];
  __shared__ float ks[8 * 256];
  __shared__ float red[256];
  __shared__ float smat[64];
  int bid = blockIdx.x;                 // b*8 + h
  int b = bid >> 3, h = bid & 7;
  int t = threadIdx.x;
  int pr = t & 63, c = pr >> 3, d = pr & 7, s = t >> 6;
  const float* qb = qkvd + (long long)b * 192 * HW_ + (long long)(h * 8) * HW_;
  const float* kb = qb + (long long)64 * HW_;
  float acc = 0.f;
  for (int n0 = 0; n0 < HW_; n0 += 256) {
    #pragma unroll
    for (int r = 0; r < 8; ++r) {
      int e = r * 256 + t;
      int row = e >> 8, n = e & 255;
      qs[row * 256 + n] = qb[(long long)row * HW_ + n0 + n];
      ks[row * 256 + n] = kb[(long long)row * HW_ + n0 + n];
    }
    __syncthreads();
    const float* qr = &qs[c * 256 + s * 64];
    const float* kr = &ks[d * 256 + s * 64];
    #pragma unroll 8
    for (int n = 0; n < 64; ++n) acc += qr[n] * kr[n];
    __syncthreads();
  }
  red[t] = acc; __syncthreads();
  if (s == 0) {
    float tot = red[pr] + red[64 + pr] + red[128 + pr] + red[192 + pr];
    smat[pr] = tot * inv[b * 64 + h * 8 + c] * inv[256 + b * 64 + h * 8 + d] * temp[h];
  }
  __syncthreads();
  if (t < 8) {
    float m = -1e30f;
    for (int dd = 0; dd < 8; ++dd) m = fmaxf(m, smat[t * 8 + dd]);
    float e[8]; float sum = 0.f;
    for (int dd = 0; dd < 8; ++dd) { e[dd] = __expf(smat[t * 8 + dd] - m); sum += e[dd]; }
    for (int dd = 0; dd < 8; ++dd) attn[bid * 64 + t * 8 + dd] = e[dd] / sum;
  }
}

// --------------------------- out = attn @ (chia * v) -----------------------
__global__ __launch_bounds__(256)
void attnout_k(const float* __restrict__ qkvd, const float* __restrict__ attn,
               const float* __restrict__ chia, float* __restrict__ cat)
{
  __shared__ float am[64];
  __shared__ float sc[8];
  int t = threadIdx.x;
  int h = blockIdx.y, b = blockIdx.z;
  if (t < 64) am[t] = attn[(b * 8 + h) * 64 + t];
  if (t < 8)  sc[t] = chia[b * 64 + h * 8 + t];
  __syncthreads();
  int pix = blockIdx.x * 256 + t;
  const float* vb = qkvd + (long long)b * 192 * HW_ + (long long)(128 + h * 8) * HW_ + pix;
  float vv[8];
  #pragma unroll
  for (int d = 0; d < 8; ++d) vv[d] = vb[(long long)d * HW_] * sc[d];
  float* ob = cat + (long long)b * 128 * HW_ + (long long)(64 + h * 8) * HW_ + pix;
  #pragma unroll
  for (int c = 0; c < 8; ++c) {
    float o = 0.f;
    #pragma unroll
    for (int d = 0; d < 8; ++d) o += am[c * 8 + d] * vv[d];
    ob[(long long)c * HW_] = o;
  }
}

// --------------------------- si: 16 -> 1, sigmoid --------------------------
__global__ __launch_bounds__(256)
void si2_k(const float* __restrict__ si16, const float* __restrict__ w2,
           const float* __restrict__ b2, float* __restrict__ spia)
{
  long long idx = (long long)blockIdx.x * 256 + threadIdx.x;
  if (idx >= (long long)B_ * HW_) return;
  long long b = idx / HW_, pix = idx % HW_;
  const float* ip = si16 + b * 16 * HW_ + pix;
  float s = b2[0];
  #pragma unroll
  for (int j = 0; j < 16; ++j) s += w2[j] * ip[(long long)j * HW_];
  spia[idx] = sigmoid_f(s);
}

// --------------------------- SimpleGate: x1 * x2 ---------------------------
__global__ __launch_bounds__(256)
void sgate_k(const float* __restrict__ f1, float* __restrict__ sg)
{
  long long idx = (long long)blockIdx.x * 256 + threadIdx.x;
  long long per = (long long)128 * HW_;
  if (idx >= (long long)B_ * per) return;
  long long b = idx / per, r = idx % per;
  sg[idx] = f1[b * 256 * HW_ + r] * f1[b * 256 * HW_ + 128 * HW_ + r];
}

// ===========================================================================
extern "C" void kernel_launch(void* const* d_in, const int* in_sizes, int n_in,
                              void* d_out, int out_size, void* d_ws, size_t ws_size,
                              hipStream_t stream)
{
  (void)in_sizes; (void)n_in; (void)out_size; (void)ws_size;
  const float* inp   = (const float*)d_in[0];
  const float* p1_w  = (const float*)d_in[1];
  const float* p1_b  = (const float*)d_in[2];
  const float* n1_g  = (const float*)d_in[3];
  const float* n1_b  = (const float*)d_in[4];
  const float* dw_w  = (const float*)d_in[5];
  const float* dw_b  = (const float*)d_in[6];
  const float* ci_w1 = (const float*)d_in[7];
  const float* ci_b1 = (const float*)d_in[8];
  const float* ci_w2 = (const float*)d_in[9];
  const float* ci_b2 = (const float*)d_in[10];
  const float* pc_w  = (const float*)d_in[11];
  const float* pc_b  = (const float*)d_in[12];
  const float* qkv_w = (const float*)d_in[13];
  const float* qkv_b = (const float*)d_in[14];
  const float* qdw_w = (const float*)d_in[15];
  const float* qdw_b = (const float*)d_in[16];
  const float* temp  = (const float*)d_in[17];
  const float* si_w1 = (const float*)d_in[18];
  const float* si_b1 = (const float*)d_in[19];
  const float* si_w2 = (const float*)d_in[20];
  const float* si_b2 = (const float*)d_in[21];
  const float* cat_w = (const float*)d_in[22];
  const float* cat_b = (const float*)d_in[23];
  const float* n2_g  = (const float*)d_in[24];
  const float* n2_b  = (const float*)d_in[25];
  const float* f1_w  = (const float*)d_in[26];
  const float* f1_b  = (const float*)d_in[27];
  const float* f2_w  = (const float*)d_in[28];
  const float* f2_b  = (const float*)d_in[29];

  float* ws = (float*)d_ws;
  const long long HWl = HW_;
  const long long PIX = (long long)B_ * HW_;          // 147456

  // workspace layout (floats), with lifetime-based reuse
  float* x    = ws;                                   // 128ch: x, later y
  float* cnn  = x    + 128 * PIX;                     // 64ch
  float* qkv  = cnn  +  64 * PIX;                     // 192ch  } 384ch region
  float* qkvd = qkv  + 192 * PIX;                     // 192ch  } -> t + f1
  float* cat  = qkvd + 192 * PIX;                     // 128ch: cat, later sg
  float* si16 = cat  + 128 * PIX;                     // 16ch
  float* spia = si16 +  16 * PIX;                     // PIX
  float* pooled = spia + PIX;                         // 256
  float* chia = pooled + 256;                         // 256
  float* nrm  = chia + 256;                           // 512
  float* attn = nrm + 512;                            // 2048
  float* y    = x;
  float* t    = qkv;                                  // 128ch (qkv dead)
  float* f1   = qkv + 128 * PIX;                      // 256ch (qkv+qkvd dead)
  float* sg   = cat;                                  // 128ch (cat dead)

  dim3 blk(256);
  auto g1 = [](long long n) { return dim3((unsigned)((n + 255) / 256)); };

  // 1) p1 conv1x1 128->128 (raw, LN needs pre-activation): 128x64 blocks
  gemm1x1_wmma<4><<<dim3(HW_/64, B_, 1), blk, 0, stream>>>(
      inp, 128*HWl, p1_w, p1_b, x, 128*HWl, 128, 2, 4, 0, nullptr, 0);
  // 2) LN + GELU (in place; per-pixel ownership)
  ln2d_k<<<g1(PIX), blk, 0, stream>>>(x, x, n1_g, n1_b, 128, 1);
  // 3) conv branch: depthwise 3x3 + GELU on ch 0-63
  dwconv3x3_k<<<g1((long long)B_*64*HWl), blk, 0, stream>>>(
      x, 128*HWl, dw_w, dw_b, cnn, 64*HWl, 64, 1);
  // 4) squeeze: global mean pool, 5) excite MLP -> chia
  pool_k<<<dim3(256), blk, 0, stream>>>(cnn, pooled);
  se_k<<<dim3(1), blk, 0, stream>>>(pooled, ci_w1, ci_b1, ci_w2, ci_b2, chia);
  // 6) qkv conv1x1 64->192 on attn_in (ch 64-127 of x): 64x128 blocks, z=3
  gemm1x1_wmma<4><<<dim3(HW_/128, B_, 3), blk, 0, stream>>>(
      x + 64*HWl, 128*HWl, qkv_w, qkv_b, qkv, 192*HWl, 64, 1, 8, 0, nullptr, 0);
  // 7) depthwise 3x3 on 192ch
  dwconv3x3_k<<<g1((long long)B_*192*HWl), blk, 0, stream>>>(
      qkv, 192*HWl, qdw_w, qdw_b, qkvd, 192*HWl, 192, 0);
  // 8) L2 norms of q,k rows; 9) 8x8 channel attention + softmax
  norm_k<<<dim3(512), blk, 0, stream>>>(qkvd, nrm);
  attn_k<<<dim3(32), blk, 0, stream>>>(qkvd, nrm, temp, attn);
  // 10) attn @ (chia*v) -> cat ch 64-127
  attnout_k<<<dim3(HW_/256, 8, B_), blk, 0, stream>>>(qkvd, attn, chia, cat);
  // 11) si1 conv1x1 64->16 + GELU on attn output: 16x128 blocks
  gemm1x1_wmma<1><<<dim3(HW_/128, B_, 1), blk, 0, stream>>>(
      cat + 64*HWl, 128*HWl, si_w1, si_b1, si16, 16*HWl, 64, 1, 8, 1, nullptr, 0);
  // 12) si2 16->1 + sigmoid -> spatial gate
  si2_k<<<g1(PIX), blk, 0, stream>>>(si16, si_w2, si_b2, spia);
  // 13) pc conv1x1 64->64, fused * spia -> cat ch 0-63: 64x128 blocks
  gemm1x1_wmma<4><<<dim3(HW_/128, B_, 1), blk, 0, stream>>>(
      cnn, 64*HWl, pc_w, pc_b, cat, 128*HWl, 64, 1, 8, 4, spia, HWl);
  // 14) cat conv1x1 128->128, fused GELU + residual(inp) -> y
  gemm1x1_wmma<4><<<dim3(HW_/64, B_, 1), blk, 0, stream>>>(
      cat, 128*HWl, cat_w, cat_b, y, 128*HWl, 128, 2, 4, 2, inp, 128*HWl);
  // 15) LN -> t
  ln2d_k<<<g1(PIX), blk, 0, stream>>>(y, t, n2_g, n2_b, 128, 0);
  // 16) f1 conv1x1 128->256: z=2 blocks of 128 rows
  gemm1x1_wmma<4><<<dim3(HW_/64, B_, 2), blk, 0, stream>>>(
      t, 128*HWl, f1_w, f1_b, f1, 256*HWl, 128, 2, 4, 0, nullptr, 0);
  // 17) SimpleGate
  sgate_k<<<g1((long long)B_*128*HWl), blk, 0, stream>>>(f1, sg);
  // 18) f2 conv1x1 128->128, fused residual(y) -> d_out
  gemm1x1_wmma<4><<<dim3(HW_/64, B_, 1), blk, 0, stream>>>(
      sg, 128*HWl, f2_w, f2_b, (float*)d_out, 128*HWl, 128, 2, 4, 3, y, 128*HWl);
}
